// RealNVP_2D_23321672417863
// MI455X (gfx1250) — matrix operand order repaired and checked by reading, not compiled, not run on hardware
//
#include <hip/hip_runtime.h>

typedef __attribute__((ext_vector_type(16))) _Float16 v16h;
typedef __attribute__((ext_vector_type(8)))  _Float16 v8h;
typedef __attribute__((ext_vector_type(8)))  float    v8f;

#define WAVES 4
#define BLOCK 128
#define HID   64
#define NL    8

// offsets (in floats) into the per-net small-weight LDS arena
#define W1OFF 0
#define B1OFF 128
#define B2OFF 192
#define B3OFF 256
#define W4OFF 320
#define B4OFF 448
#define SMALLSZ 452

__global__ __launch_bounds__(BLOCK) void realnvp2d_kernel(
    const float* __restrict__ x,
    const float* __restrict__ s_w1, const float* __restrict__ s_b1,
    const float* __restrict__ s_w2, const float* __restrict__ s_b2,
    const float* __restrict__ s_w3, const float* __restrict__ s_b3,
    const float* __restrict__ s_w4, const float* __restrict__ s_b4,
    const float* __restrict__ t_w1, const float* __restrict__ t_b1,
    const float* __restrict__ t_w2, const float* __restrict__ t_b2,
    const float* __restrict__ t_w3, const float* __restrict__ t_b3,
    const float* __restrict__ t_w4, const float* __restrict__ t_b4,
    float* __restrict__ out, int n)
{
    // [net*2 + mat] : W2/W3 for both nets of the current coupling layer, f16 row-major
    __shared__ __align__(32) _Float16 sW[4][HID * HID];
    __shared__ float sSmall[2][SMALLSZ];                   // w1,b1,b2,b3,w4,b4 per net (f32)
    __shared__ __align__(32) _Float16 sH[WAVES][32 * HID]; // activations [sample][feat]

    const int tid  = threadIdx.x;
    const int lane = tid & 31;
    const int wave = tid >> 5;
    const int hi   = lane >> 4;    // half-wave (fragment K/M split)
    const int l16  = lane & 15;

    const int gid  = blockIdx.x * BLOCK + wave * 32 + lane;
    const int gidc = gid < n ? gid : n - 1;

    float y0 = x[2 * gidc + 0];
    float y1 = x[2 * gidc + 1];
    float ld = 0.f;

    const float* bigsrc[4] = { s_w2, s_w3, t_w2, t_w3 };

    for (int l = 0; l < NL; ++l) {
        const int a = (l & 1);     // conditioning component
        const int b = a ^ 1;       // transformed component

        __syncthreads();
        // ---- cooperative stage of this layer's weights (f32 -> f16 for W2/W3) ----
        #pragma unroll
        for (int m4 = 0; m4 < 4; ++m4) {
            const float4* src = (const float4*)(bigsrc[m4] + l * HID * HID);
            _Float16* dst = &sW[m4][0];
            #pragma unroll
            for (int i = tid; i < (HID * HID) / 4; i += BLOCK) {
                float4 v = src[i];
                dst[4 * i + 0] = (_Float16)v.x;
                dst[4 * i + 1] = (_Float16)v.y;
                dst[4 * i + 2] = (_Float16)v.z;
                dst[4 * i + 3] = (_Float16)v.w;
            }
        }
        {
            // 128-wide pieces
            sSmall[0][W1OFF + tid] = s_w1[l * 128 + tid];
            sSmall[1][W1OFF + tid] = t_w1[l * 128 + tid];
            sSmall[0][W4OFF + tid] = s_w4[l * 128 + tid];
            sSmall[1][W4OFF + tid] = t_w4[l * 128 + tid];
            if (tid < 64) {
                sSmall[0][B1OFF + tid] = s_b1[l * 64 + tid];
                sSmall[1][B1OFF + tid] = t_b1[l * 64 + tid];
                sSmall[0][B2OFF + tid] = s_b2[l * 64 + tid];
                sSmall[1][B2OFF + tid] = t_b2[l * 64 + tid];
                sSmall[0][B3OFF + tid] = s_b3[l * 64 + tid];
                sSmall[1][B3OFF + tid] = t_b3[l * 64 + tid];
            }
            if (tid < 2) {
                sSmall[0][B4OFF + tid] = s_b4[l * 2 + tid];
                sSmall[1][B4OFF + tid] = t_b4[l * 2 + tid];
            }
        }
        // prefetch next layer's big weights while we compute (global_prefetch_b8)
        if (l + 1 < NL) {
            #pragma unroll
            for (int m4 = 0; m4 < 4; ++m4)
                __builtin_prefetch(bigsrc[m4] + (l + 1) * HID * HID + tid * 32, 0, 0);
        }
        __syncthreads();

        float outv[2];
        #pragma unroll
        for (int net = 0; net < 2; ++net) {
            const float* sm = sSmall[net];
            const float xa = a ? y1 : y0;
            _Float16* hrow = &sH[wave][lane * HID];

            // ---- layer 1: rank-1 (masked input has one live scalar), per-lane VALU ----
            #pragma unroll
            for (int g = 0; g < 8; ++g) {
                v8h hv;
                #pragma unroll
                for (int j = 0; j < 8; ++j) {
                    const int f = 8 * g + j;
                    float v = sm[W1OFF + 2 * f + a] * xa + sm[B1OFF + f];
                    hv[j] = (_Float16)fmaxf(v, 0.f);
                }
                *(v8h*)(hrow + 8 * g) = hv;
            }

            // ---- layers 2 and 3: 64x64 via v_wmma_f32_16x16x32_f16 ----
            // A-fragment is loaded once per (mt,kk) and reused for BOTH 16-sample
            // subtiles (two independent WMMAs per LDS load).
            #pragma unroll
            for (int mat = 0; mat < 2; ++mat) {
                const _Float16* W = &sW[net * 2 + mat][0];
                const int boff = mat ? B3OFF : B2OFF;

                // preload B fragments for both subtiles (sample = lane column)
                v16h bf[2][2];
                #pragma unroll
                for (int st = 0; st < 2; ++st) {
                    const int srow = st * 16 + l16;
                    bf[st][0] = *(const v16h*)&sH[wave][srow * HID +      hi * 16];
                    bf[st][1] = *(const v16h*)&sH[wave][srow * HID + 32 + hi * 16];
                }

                v8f c[4][2];
                #pragma unroll
                for (int mt = 0; mt < 4; ++mt) {
                    const int fb = mt * 16 + hi * 8;    // feature base in C fragment
                    v8f binit;
                    #pragma unroll
                    for (int v = 0; v < 8; ++v) binit[v] = sm[boff + fb + v];
                    c[mt][0] = binit;
                    c[mt][1] = binit;
                    #pragma unroll
                    for (int kk = 0; kk < 2; ++kk) {
                        const _Float16* wr = &W[(mt * 16 + l16) * HID + kk * 32 + hi * 16];
                        v8h alo = *(const v8h*)(wr);        // K 0..7 of this half
                        v8h ahi = *(const v8h*)(wr + 8);    // K 8..15 of this half
                        v16h av;
                        #pragma unroll
                        for (int i = 0; i < 8; ++i) { av[i] = alo[i]; av[i + 8] = ahi[i]; }
                        c[mt][0] = __builtin_amdgcn_wmma_f32_16x16x32_f16(
                            false, av, false, bf[0][kk], (short)0, c[mt][0], false, false);
                        c[mt][1] = __builtin_amdgcn_wmma_f32_16x16x32_f16(
                            false, av, false, bf[1][kk], (short)0, c[mt][1], false, false);
                    }
                }

                // ReLU + writeback (all B fragments already consumed)
                #pragma unroll
                for (int st = 0; st < 2; ++st) {
                    const int srow = st * 16 + l16;
                    #pragma unroll
                    for (int mt = 0; mt < 4; ++mt) {
                        const int fb = mt * 16 + hi * 8;
                        v8h o;
                        #pragma unroll
                        for (int v = 0; v < 8; ++v)
                            o[v] = (_Float16)fmaxf(c[mt][st][v], 0.f);
                        *(v8h*)&sH[wave][srow * HID + fb] = o;
                    }
                }
            }

            // ---- layer 4: only row b of w4 matters; per-lane 64-FMA dot ----
            float acc = sm[B4OFF + b];
            #pragma unroll
            for (int g = 0; g < 8; ++g) {
                v8h hv = *(const v8h*)(hrow + 8 * g);
                #pragma unroll
                for (int j = 0; j < 8; ++j)
                    acc += sm[W4OFF + b * HID + 8 * g + j] * (float)hv[j];
            }
            outv[net] = acc;
        }

        // ---- coupling update: y_b = y_b*exp(tanh(s_out)) + t_out ; ld += tanh(s_out)
        const float e2 = __expf(2.f * outv[0]);
        const float s  = (e2 - 1.f) / (e2 + 1.f);      // tanh
        const float t  = outv[1];
        if (b == 0) y0 = y0 * __expf(s) + t;
        else        y1 = y1 * __expf(s) + t;
        ld += s;
    }

    if (gid < n) {
        out[2 * gid + 0] = y0;
        out[2 * gid + 1] = y1;
        out[2 * n + gid] = ld;
    }
}

extern "C" void kernel_launch(void* const* d_in, const int* in_sizes, int n_in,
                              void* d_out, int out_size, void* d_ws, size_t ws_size,
                              hipStream_t stream) {
    const float* x    = (const float*)d_in[0];
    const float* sw1  = (const float*)d_in[1];
    const float* sb1  = (const float*)d_in[2];
    const float* sw2  = (const float*)d_in[3];
    const float* sb2  = (const float*)d_in[4];
    const float* sw3  = (const float*)d_in[5];
    const float* sb3  = (const float*)d_in[6];
    const float* sw4  = (const float*)d_in[7];
    const float* sb4  = (const float*)d_in[8];
    const float* tw1  = (const float*)d_in[9];
    const float* tb1  = (const float*)d_in[10];
    const float* tw2  = (const float*)d_in[11];
    const float* tb2  = (const float*)d_in[12];
    const float* tw3  = (const float*)d_in[13];
    const float* tb3  = (const float*)d_in[14];
    const float* tw4  = (const float*)d_in[15];
    const float* tb4  = (const float*)d_in[16];

    const int n = in_sizes[0] / 2;   // N samples (x is N x 2)
    const int grid = (n + BLOCK - 1) / BLOCK;

    realnvp2d_kernel<<<grid, BLOCK, 0, stream>>>(
        x, sw1, sb1, sw2, sb2, sw3, sb3, sw4, sb4,
        tw1, tb1, tw2, tb2, tw3, tb3, tw4, tb4,
        (float*)d_out, n);
}
